// SpatialGCM_69054484185237
// MI455X (gfx1250) — matrix operand order repaired and checked by reading, not compiled, not run on hardware
//
#include <hip/hip_runtime.h>
#include <hip/hip_bf16.h>

typedef __attribute__((ext_vector_type(16))) _Float16 v16h_t;
typedef __attribute__((ext_vector_type(8)))  _Float16 v8h_t;
typedef __attribute__((ext_vector_type(8)))  float    v8f_t;

#define NB 8          // query batch
#define NM 32         // memory bank size
#define NC 512        // channels
#define HH 64
#define WW 64
#define NPIX (HH*WW)  // 4096

// ---------------------------------------------------------------------------
// Phase 1a: per-pixel inverse L2 norms for query (8 imgs) and memory (32 imgs)
// ---------------------------------------------------------------------------
__global__ __launch_bounds__(256) void norms_k(const float* __restrict__ q,
                                               const float* __restrict__ m,
                                               float* __restrict__ invq,
                                               float* __restrict__ invm) {
    int t = blockIdx.x * blockDim.x + threadIdx.x;   // over (8+32)*4096 pixels
    int img = t >> 12;
    int hw  = t & (NPIX - 1);
    const float* src;
    float* dst;
    if (img < NB) { src = q + (size_t)img * NC * NPIX + hw; dst = invq + t; }
    else { int mi = img - NB; src = m + (size_t)mi * NC * NPIX + hw; dst = invm + mi * NPIX + hw; }
    float s = 0.f;
#pragma unroll 8
    for (int c = 0; c < NC; ++c) { float x = src[(size_t)c * NPIX]; s = fmaf(x, x, s); }
    *dst = 1.0f / fmaxf(sqrtf(s), 1e-12f);
}

// ---------------------------------------------------------------------------
// Phase 1b: normalize + fp32->f16 + NCHW -> [b][hw][c] transpose (query)
// LDS-tiled 32x32 so both global read and write are coalesced.
// ---------------------------------------------------------------------------
__global__ __launch_bounds__(256) void ntq_k(const float* __restrict__ q,
                                             const float* __restrict__ invq,
                                             _Float16* __restrict__ qh) {
    __shared__ float tile[32][33];
    __shared__ float invs[32];
    const int b   = blockIdx.z;
    const int c0  = blockIdx.y * 32;
    const int hw0 = blockIdx.x * 32;
    for (int i = threadIdx.y; i < 32; i += 8)
        tile[i][threadIdx.x] = q[((size_t)b * NC + c0 + i) * NPIX + hw0 + threadIdx.x];
    if (threadIdx.y == 0) invs[threadIdx.x] = invq[b * NPIX + hw0 + threadIdx.x];
    __syncthreads();
    for (int i = threadIdx.y; i < 32; i += 8)
        qh[((size_t)b * NPIX + hw0 + i) * NC + c0 + threadIdx.x] =
            (_Float16)(tile[threadIdx.x][i] * invs[i]);
}

// memory bank: NCHW -> [hw][m][c] (channels innermost, per-m contiguous)
__global__ __launch_bounds__(256) void ntm_k(const float* __restrict__ m,
                                             const float* __restrict__ invm,
                                             _Float16* __restrict__ mh) {
    __shared__ float tile[32][33];
    __shared__ float invs[32];
    const int mi  = blockIdx.z;
    const int c0  = blockIdx.y * 32;
    const int hw0 = blockIdx.x * 32;
    for (int i = threadIdx.y; i < 32; i += 8)
        tile[i][threadIdx.x] = m[((size_t)mi * NC + c0 + i) * NPIX + hw0 + threadIdx.x];
    if (threadIdx.y == 0) invs[threadIdx.x] = invm[mi * NPIX + hw0 + threadIdx.x];
    __syncthreads();
    for (int i = threadIdx.y; i < 32; i += 8)
        mh[((size_t)(hw0 + i) * NM + mi) * NC + c0 + threadIdx.x] =
            (_Float16)(tile[threadIdx.x][i] * invs[i]);
}

// ---------------------------------------------------------------------------
// Phase 2: WMMA similarity. One wave owns a query-pixel pair (w, w+1) x all 8
// batches (M=16 rows), loops 2 m-tiles (N=16) x 3 row-shifts x 4 memory
// columns, K=512 in 16 steps of v_wmma_f32_16x16x32_f16. Running max folds
// the 3x3 neighborhood; relu is free (max init = 0); mean over m via shfl.
// ---------------------------------------------------------------------------
__global__ __launch_bounds__(256) void sim_k(const _Float16* __restrict__ qh,
                                             const _Float16* __restrict__ mh,
                                             float* __restrict__ out) {
    const int h    = blockIdx.y;
    const int wt   = blockIdx.x;
    const int wave = threadIdx.x >> 5;
    const int lane = threadIdx.x & 31;
    const int half = lane >> 4;       // lane half: selects K sub-runs & C rows 8..15
    const int lm   = lane & 15;
    const int wq0  = wt * 16 + wave * 2;           // first query pixel of the pair

    // A-matrix row for this lane: row M = lm -> (b = M&7, px = M>>3)
    const int ab  = lm & 7;
    const int apx = lm >> 3;
    const _Float16* abase = qh + (((size_t)ab * HH + h) * WW + (wq0 + apx)) * NC;

    float sum[8] = {0.f, 0.f, 0.f, 0.f, 0.f, 0.f, 0.f, 0.f};

    for (int mt = 0; mt < 2; ++mt) {
        const _Float16* mtbase = mh + (size_t)(mt * 16 + lm) * NC;  // B col N = lm
        float maxv[8] = {0.f, 0.f, 0.f, 0.f, 0.f, 0.f, 0.f, 0.f};  // init 0 == fused relu

        for (int di = 0; di < 3; ++di) {
            const int hm = h + di - 1;
            if (hm < 0 || hm >= HH) continue;            // wave-uniform skip
            for (int wi = 0; wi < 4; ++wi) {
                const int wm = wq0 - 1 + wi;
                if (wm < 0 || wm >= WW) continue;        // wave-uniform skip
                const _Float16* bbase = mtbase + (size_t)(hm * WW + wm) * NM * NC;

                v8f_t acc = {0.f, 0.f, 0.f, 0.f, 0.f, 0.f, 0.f, 0.f};
#pragma unroll
                for (int k = 0; k < 16; ++k) {
                    // A lane layout: K runs [8*half, +8) and [16+8*half, +8)
                    const int koa = k * 32 + half * 8;
                    v8h_t alo = *(const v8h_t*)(abase + koa);
                    v8h_t ahi = *(const v8h_t*)(abase + koa + 16);
                    v16h_t av = __builtin_shufflevector(alo, ahi,
                        0,1,2,3,4,5,6,7,8,9,10,11,12,13,14,15);
                    // B lane layout: K runs [8*half, +8) and [16+8*half, +8)
                    const int kob = k * 32 + half * 8;
                    v8h_t blo = *(const v8h_t*)(bbase + kob);
                    v8h_t bhi = *(const v8h_t*)(bbase + kob + 16);
                    v16h_t bv = __builtin_shufflevector(blo, bhi,
                        0,1,2,3,4,5,6,7,8,9,10,11,12,13,14,15);
                    acc = __builtin_amdgcn_wmma_f32_16x16x32_f16(
                        false, av, false, bv, (short)0, acc, false, false);
                }
                // Fold into running max. C/D row for VGPR r is M = r + 8*half,
                // whose query pixel is wq0 + half -> validity uniform per half.
                const int wqrow = wq0 + half;
                const bool valid = (wm >= wqrow - 1) && (wm <= wqrow + 1);
#pragma unroll
                for (int r = 0; r < 8; ++r) {
                    float mx = fmaxf(maxv[r], acc[r]);
                    maxv[r] = valid ? mx : maxv[r];
                }
            }
        }
#pragma unroll
        for (int r = 0; r < 8; ++r) sum[r] += maxv[r];   // accumulate over m-tiles
    }

    // Mean over m: reduce across the 16 lanes of each half (xor of bits 0..3
    // stays within a half on wave32).
#pragma unroll
    for (int r = 0; r < 8; ++r) {
        float v = sum[r];
        v += __shfl_xor(v, 1, 32);
        v += __shfl_xor(v, 2, 32);
        v += __shfl_xor(v, 4, 32);
        v += __shfl_xor(v, 8, 32);
        sum[r] = v;
    }
    if (lm == 0) {
        const int wq = wq0 + half;     // half 0 -> pixel w, half 1 -> pixel w+1
#pragma unroll
        for (int r = 0; r < 8; ++r)    // row M = r + 8*half -> b = r
            out[(size_t)r * NPIX + h * WW + wq] = 1.0f - sum[r] * (1.0f / 32.0f);
    }
}

// ---------------------------------------------------------------------------
extern "C" void kernel_launch(void* const* d_in, const int* in_sizes, int n_in,
                              void* d_out, int out_size, void* d_ws, size_t ws_size,
                              hipStream_t stream) {
    (void)in_sizes; (void)n_in; (void)out_size; (void)ws_size;
    const float* q = (const float*)d_in[0];   // (8, 512, 64, 64) fp32
    const float* m = (const float*)d_in[1];   // (32, 512, 64, 64) fp32
    float* out = (float*)d_out;               // (8, 1, 64, 64) fp32

    // workspace layout
    char* w = (char*)d_ws;
    float*    invq = (float*)(w);                                  // 128 KB
    float*    invm = (float*)(w + 131072);                         // 512 KB
    _Float16* qh   = (_Float16*)(w + (1u << 20));                  // 32 MB
    _Float16* mh   = (_Float16*)(w + (1u << 20) + (32u << 20));    // 128 MB

    norms_k<<<(NB + NM) * NPIX / 256, 256, 0, stream>>>(q, m, invq, invm);

    dim3 tb(32, 8);
    ntq_k<<<dim3(NPIX / 32, NC / 32, NB), tb, 0, stream>>>(q, invq, qh);
    ntm_k<<<dim3(NPIX / 32, NC / 32, NM), tb, 0, stream>>>(m, invm, mh);

    sim_k<<<dim3(WW / 16, HH), 256, 0, stream>>>(qh, mh, out);
}
